// LSTMGCNModel_3891240370260
// MI455X (gfx1250) — matrix operand order repaired and checked by reading, not compiled, not run on hardware
//
#include <hip/hip_runtime.h>

// ---------------------------------------------------------------------------
// GCLSTM (single step, H=C=0) fused kernel for gfx1250 (MI455X).
// Reduces to: I=sig(xW0+b0), T=tanh(xW2+b2), C=I*T, O=sig(xW3+b3+wc2*C),
//             h=relu(O*tanh(C)), out=h@linW+lin_b.
// GEMMs via v_wmma_f32_16x16x32_bf16 with bf16 hi/lo truncation split
// (3 products: hi*hi + hi*lo + lo*hi) for ~fp32 accuracy.
// Memory-bound: ~38MB HBM traffic @ 23.3 TB/s ~= 1.7us floor; each wave
// processes 2 row-tiles to halve L2 weight-fragment traffic.
// ---------------------------------------------------------------------------

typedef __attribute__((ext_vector_type(16))) __bf16         v16bf;
typedef __attribute__((ext_vector_type(8)))  float          v8f;
typedef __attribute__((ext_vector_type(8)))  unsigned int   v8u32;
typedef __attribute__((ext_vector_type(16))) unsigned short v16u16;
typedef __attribute__((ext_vector_type(8)))  unsigned short v8u16;

#define NNODES   50000
#define MTILES   3125          // 50000 / 16
#define M2       2             // row-tiles per wave (B-fragment reuse)
#define WAVES_PB 2

// workspace layout (bytes)
#define OFF_BG_HI   0u          // 3 gates * 16384 u16 = 98304 B
#define OFF_BG_LO   98304u
#define OFF_BL_HI   196608u     // 8192 u16 = 16384 B
#define OFF_BL_LO   212992u
#define OFF_BETA    229376u     // 4*128 float = 2048 B

__device__ __forceinline__ unsigned fbits(float f) {
    return __builtin_bit_cast(unsigned, f);
}
__device__ __forceinline__ float fcast(unsigned u) {
    return __builtin_bit_cast(float, u);
}
__device__ __forceinline__ float hi_f(float f) {          // bf16-truncated value
    return fcast(fbits(f) & 0xffff0000u);
}
// pack truncated bf16 of (even,odd) floats -> dword {bf(odd)<<16 | bf(even)}
__device__ __forceinline__ unsigned pack_bf(float e, float o) {
    return __builtin_amdgcn_perm(fbits(o), fbits(e), 0x07060302u);
}
__device__ __forceinline__ float sigm_(float x) {
    return 1.0f / (1.0f + __expf(-x));
}
__device__ __forceinline__ float tanh_(float x) {
    x = fminf(fmaxf(x, -15.0f), 15.0f);
    float e = __expf(2.0f * x);
    return (e - 1.0f) / (e + 1.0f);
}
__device__ __forceinline__ v8f wmma_bf16(v16bf a, v16bf b, v8f c) {
    return __builtin_amdgcn_wmma_f32_16x16x32_bf16(
        false, a, false, b, (short)0, c, false, false);
}

// ---------------------------------------------------------------------------
// Prep: repack Wx{0,2,3} and lin_W into B-operand (32x16 bf16) fragment
// layout, hi/lo planes (truncation split); fuse bias vectors.
// Fragment layout: element index = ((nt*4+ks)*32 + lane)*16 + i
//   lane<16 : K = ks*32 + (i&7) + (i>>3)*16,     N = nt*16 + lane
//   lane>=16: K = ks*32 + 8 + (i&7) + (i>>3)*16, N = nt*16 + (lane-16)
// ---------------------------------------------------------------------------
__global__ void gclstm_pack(const float* __restrict__ Wx,
                            const float* __restrict__ b_gate,
                            const float* __restrict__ w_c,
                            const float* __restrict__ cheb_b,
                            const float* __restrict__ lin_W,
                            unsigned short* __restrict__ Bg_hi,
                            unsigned short* __restrict__ Bg_lo,
                            unsigned short* __restrict__ Bl_hi,
                            unsigned short* __restrict__ Bl_lo,
                            float* __restrict__ beta) {
    int tid = blockIdx.x * blockDim.x + threadIdx.x;
    const int GATE_ELEMS = 3 * 16384;
    if (tid < GATE_ELEMS) {
        int gl   = tid / 16384;          // 0,1,2 -> gates 0,2,3
        int r    = tid & 16383;
        int i    = r & 15;
        int lane = (r >> 4) & 31;
        int ks   = (r >> 9) & 3;
        int nt   = (r >> 11) & 7;
        int k    = ks * 32 + (lane >> 4) * 8 + (i & 7) + (i >> 3) * 16;
        int n    = nt * 16 + (lane & 15);
        int g    = (gl == 0) ? 0 : gl + 1;
        float w  = Wx[g * 16384 + k * 128 + n];
        Bg_hi[tid] = (unsigned short)(fbits(w) >> 16);
        Bg_lo[tid] = (unsigned short)(fbits(w - hi_f(w)) >> 16);
    } else if (tid < GATE_ELEMS + 8192) {
        int r    = tid - GATE_ELEMS;
        int i    = r & 15;
        int lane = (r >> 4) & 31;
        int ks   = (r >> 9) & 3;
        int nt   = (r >> 11) & 3;
        int k    = ks * 32 + (lane >> 4) * 8 + (i & 7) + (i >> 3) * 16;
        int n    = nt * 16 + (lane & 15);
        float w  = lin_W[k * 64 + n];
        Bl_hi[r] = (unsigned short)(fbits(w) >> 16);
        Bl_lo[r] = (unsigned short)(fbits(w - hi_f(w)) >> 16);
    } else if (tid < GATE_ELEMS + 8192 + 512) {
        int r   = tid - GATE_ELEMS - 8192;
        int row = r >> 7;                 // 0..3
        int n   = r & 127;
        float v;
        if (row < 3) {
            int g = (row == 0) ? 0 : row + 1;            // gates 0,2,3
            v = cheb_b[g * 128 + n] + b_gate[g * 128 + n];
        } else {
            v = w_c[2 * 128 + n];                         // peephole for O
        }
        beta[r] = v;
    }
}

// ---------------------------------------------------------------------------
// Main fused kernel: one wave per pair of 16-row node tiles.
// ---------------------------------------------------------------------------
__global__ __launch_bounds__(WAVES_PB * 32)
void gclstm_fused(const float* __restrict__ x,
                  const unsigned short* __restrict__ Bg_hi,
                  const unsigned short* __restrict__ Bg_lo,
                  const unsigned short* __restrict__ Bl_hi,
                  const unsigned short* __restrict__ Bl_lo,
                  const float* __restrict__ beta,
                  const float* __restrict__ lin_b,
                  float* __restrict__ out) {
    // padded row stride (136 u16 = 272B) -> conflict-free ds_load_b128
    __shared__ unsigned short ldsH[WAVES_PB][M2][2][16 * 136];

    int lane = threadIdx.x & 31;
    int wave = threadIdx.x >> 5;
    int task = blockIdx.x * WAVES_PB + wave;
    int mt0  = task * M2;
    if (mt0 >= MTILES) return;             // wave-uniform: EXEC stays all-1
    int kh   = lane >> 4;                  // lane half (0/1)
    int lrow = lane & 15;

    // ---- load + split the two 16x128 x tiles into A fragments (once) ----
    v16bf aHi[M2][4], aLo[M2][4];
#pragma unroll
    for (int t = 0; t < M2; ++t) {
        int row = mt0 * 16 + t * 16 + lrow;
        if (row >= NNODES) row = NNODES - 1;           // tail clamp (no OOB)
        const float* xr = x + (size_t)row * 128;
#pragma unroll
        for (int ks = 0; ks < 4; ++ks) {
            int c0 = ks * 32 + kh * 8;
            float4 f0 = *(const float4*)(xr + c0);
            float4 f1 = *(const float4*)(xr + c0 + 4);
            float4 f2 = *(const float4*)(xr + c0 + 16);
            float4 f3 = *(const float4*)(xr + c0 + 20);
            float av[16] = {f0.x, f0.y, f0.z, f0.w, f1.x, f1.y, f1.z, f1.w,
                            f2.x, f2.y, f2.z, f2.w, f3.x, f3.y, f3.z, f3.w};
            v8u32 ph, pl;
#pragma unroll
            for (int p = 0; p < 8; ++p) {
                float e = av[2 * p], o = av[2 * p + 1];
                ph[p] = pack_bf(e, o);                              // 1 v_perm
                pl[p] = pack_bf(e - hi_f(e), o - hi_f(o));
            }
            aHi[t][ks] = __builtin_bit_cast(v16bf, ph);
            aLo[t][ks] = __builtin_bit_cast(v16bf, pl);
        }
    }

    // ---- gate GEMMs + fused LSTM epilogue, one 16-col tile at a time ----
    for (int nt = 0; nt < 8; ++nt) {
        v8f acc[M2][3];
#pragma unroll
        for (int t = 0; t < M2; ++t)
#pragma unroll
            for (int g = 0; g < 3; ++g) acc[t][g] = (v8f){};

#pragma unroll
        for (int ks = 0; ks < 4; ++ks) {
            int fo = ((nt * 4 + ks) * 32 + lane) * 16;
#pragma unroll
            for (int g = 0; g < 3; ++g) {
                v16bf bh = __builtin_bit_cast(
                    v16bf, *(const v16u16*)(Bg_hi + g * 16384 + fo));
                v16bf bl = __builtin_bit_cast(
                    v16bf, *(const v16u16*)(Bg_lo + g * 16384 + fo));
                // interleave over t so consecutive WMMAs hit different accs
#pragma unroll
                for (int t = 0; t < M2; ++t)
                    acc[t][g] = wmma_bf16(aHi[t][ks], bh, acc[t][g]);
#pragma unroll
                for (int t = 0; t < M2; ++t)
                    acc[t][g] = wmma_bf16(aHi[t][ks], bl, acc[t][g]);
#pragma unroll
                for (int t = 0; t < M2; ++t)
                    acc[t][g] = wmma_bf16(aLo[t][ks], bh, acc[t][g]);
            }
        }

        // fused LSTM elementwise; C/D layout: lane holds col n, rows v+8*kh
        int n = nt * 16 + lrow;
        float be0 = beta[n], be2 = beta[128 + n];
        float be3 = beta[256 + n], wc2 = beta[384 + n];
#pragma unroll
        for (int t = 0; t < M2; ++t) {
            unsigned short* hHi = &ldsH[wave][t][0][0];
            unsigned short* hLo = &ldsH[wave][t][1][0];
#pragma unroll
            for (int v = 0; v < 8; ++v) {
                float I = sigm_(acc[t][0][v] + be0);
                float T = tanh_(acc[t][1][v] + be2);
                float C = I * T;
                float O = sigm_(acc[t][2][v] + be3 + wc2 * C);
                float H = fmaxf(O * tanh_(C), 0.0f);
                int row = v + kh * 8;
                hHi[row * 136 + n] = (unsigned short)(fbits(H) >> 16);
                hLo[row * 136 + n] =
                    (unsigned short)(fbits(H - hi_f(H)) >> 16);
            }
        }
    }

    // ---- out = h @ lin_W + lin_b ; A fragments re-read from LDS ----
#pragma unroll
    for (int t = 0; t < M2; ++t) {
        int mt = mt0 + t;
        if (mt >= MTILES) break;           // wave-uniform tail skip
        const unsigned short* hHi = &ldsH[wave][t][0][0];
        const unsigned short* hLo = &ldsH[wave][t][1][0];
        for (int nt = 0; nt < 4; ++nt) {
            v8f acc = {};
#pragma unroll
            for (int ks = 0; ks < 4; ++ks) {
                int c0 = ks * 32 + kh * 8;
                v8u16 h0 = *(const v8u16*)(hHi + lrow * 136 + c0);
                v8u16 h1 = *(const v8u16*)(hHi + lrow * 136 + c0 + 16);
                v8u16 l0 = *(const v8u16*)(hLo + lrow * 136 + c0);
                v8u16 l1 = *(const v8u16*)(hLo + lrow * 136 + c0 + 16);
                v16u16 ah, al;
#pragma unroll
                for (int i = 0; i < 8; ++i) {
                    ah[i] = h0[i]; ah[8 + i] = h1[i];
                    al[i] = l0[i]; al[8 + i] = l1[i];
                }
                v16bf aHi2 = __builtin_bit_cast(v16bf, ah);
                v16bf aLo2 = __builtin_bit_cast(v16bf, al);

                int fo = ((nt * 4 + ks) * 32 + lane) * 16;
                v16bf bh = __builtin_bit_cast(v16bf, *(const v16u16*)(Bl_hi + fo));
                v16bf bl = __builtin_bit_cast(v16bf, *(const v16u16*)(Bl_lo + fo));
                acc = wmma_bf16(aHi2, bh, acc);
                acc = wmma_bf16(aHi2, bl, acc);
                acc = wmma_bf16(aLo2, bh, acc);
            }
            int j = nt * 16 + lrow;
            float bias = lin_b[j];
#pragma unroll
            for (int v = 0; v < 8; ++v) {
                int m = mt * 16 + v + kh * 8;
                out[(size_t)m * 64 + j] = acc[v] + bias;  // lanes 0-15: 64B run
            }
        }
    }
}

// ---------------------------------------------------------------------------
extern "C" void kernel_launch(void* const* d_in, const int* in_sizes, int n_in,
                              void* d_out, int out_size, void* d_ws, size_t ws_size,
                              hipStream_t stream) {
    const float* x      = (const float*)d_in[0];
    // d_in[1] edge_index, d_in[2] edge_weight, d_in[6] cheb_W: provably unused
    const float* Wx     = (const float*)d_in[3];
    const float* b_gate = (const float*)d_in[4];
    const float* w_c    = (const float*)d_in[5];
    const float* cheb_b = (const float*)d_in[7];
    const float* lin_W  = (const float*)d_in[8];
    const float* lin_b  = (const float*)d_in[9];
    float* out          = (float*)d_out;

    char* ws = (char*)d_ws;
    unsigned short* Bg_hi = (unsigned short*)(ws + OFF_BG_HI);
    unsigned short* Bg_lo = (unsigned short*)(ws + OFF_BG_LO);
    unsigned short* Bl_hi = (unsigned short*)(ws + OFF_BL_HI);
    unsigned short* Bl_lo = (unsigned short*)(ws + OFF_BL_LO);
    float*          beta  = (float*)(ws + OFF_BETA);

    const int pack_threads = 3 * 16384 + 8192 + 512;       // 57856
    gclstm_pack<<<(pack_threads + 255) / 256, 256, 0, stream>>>(
        Wx, b_gate, w_c, cheb_b, lin_W, Bg_hi, Bg_lo, Bl_hi, Bl_lo, beta);

    const int tasks  = (MTILES + M2 - 1) / M2;             // 1563
    const int blocks = (tasks + WAVES_PB - 1) / WAVES_PB;  // 782
    gclstm_fused<<<blocks, WAVES_PB * 32, 0, stream>>>(
        x, Bg_hi, Bg_lo, Bl_hi, Bl_lo, beta, lin_b, out);
}
